// ReactionEncoder_20469814133015
// MI455X (gfx1250) — compile-verified, matrix operand order checked by reading, not run on hardware
//
#include <hip/hip_runtime.h>
#include <hip/hip_fp16.h>
#include <math.h>

// ReactionEncoder for MI455X (gfx1250, wave32, WMMA + TDM).
// - segment_sum commutes with the linear message transform -> DMPNN messages
//   become scatter-add (N x 320) + small GEMM (43 GFLOP -> 1.3 GFLOP, no 268MB buf).
// - Output is only row N+1 -> attention collapses to 1 query row vs K/V.
// - GEMMs: block-tiled WMMA f16 (BM=128,BN=64,BK=32), panels staged in LDS by the
//   Tensor Data Mover (tensor_load_to_lds, double-buffered, s_wait_tensorcnt).
// Roofline: edge scatters dominate (~4.3 GB -> ~190us @ 23.3 TB/s); GEMMs ~36 GFLOP.

#define N_NODES 8192
#define N_EDGES 262144
#define HID     256
#define ED      64
#define CATK    320
#define NN      (N_NODES + 2)
#define NPAD    8208          // NN padded to multiple of 16

typedef __attribute__((ext_vector_type(16))) _Float16 v16h;
typedef __attribute__((ext_vector_type(8)))  float    v8f;
typedef __attribute__((ext_vector_type(4)))  unsigned int u32x4;
typedef __attribute__((ext_vector_type(8)))  int      i32x8;
typedef __attribute__((ext_vector_type(4)))  int      i32x4;

#if defined(__HIP_DEVICE_COMPILE__) && defined(__has_builtin)
#if __has_builtin(__builtin_amdgcn_tensor_load_to_lds) && __has_builtin(__builtin_amdgcn_s_wait_tensorcnt)
#define HAVE_TDM 1
#endif
#endif

__device__ __forceinline__ float sigmoidf_(float x){ return 1.0f/(1.0f+__expf(-x)); }

// ---------------- elementwise helpers ----------------
__global__ void k_zero(float* p, int n){
  int i = blockIdx.x*blockDim.x + threadIdx.x;
  if (i < n) p[i] = 0.0f;
}
__global__ void k_add(const float* a, const float* b, float* o, int n){
  int i = blockIdx.x*blockDim.x + threadIdx.x;
  if (i < n) o[i] = a[i] + b[i];
}
__global__ void k_copy_cols(const float* in, int ldi, float* out, int ldo, int coff,
                            int rows, int cols){
  int i = blockIdx.x*blockDim.x + threadIdx.x;
  if (i >= rows*cols) return;
  int r = i / cols, c = i - r*cols;
  out[(size_t)r*ldo + coff + c] = in[(size_t)r*ldi + c];
}

// ---------------- edge scatter (segment_sum) ----------------
__global__ void k_scatter_edge64(const float* __restrict__ ea, const int* __restrict__ dst,
                                 float* out, int ldo){
  long long i = (long long)blockIdx.x*blockDim.x + threadIdx.x;
  if (i >= (long long)N_EDGES*ED) return;
  int e = (int)(i >> 6);
  int d = (int)(i & 63);
  atomicAdd(&out[(size_t)dst[e]*ldo + d], ea[(size_t)e*ED + d]);
}
__global__ void k_scatter_node256(const float* __restrict__ h, const int* __restrict__ src,
                                  const int* __restrict__ dst, float* out, int ldo){
  long long i = (long long)blockIdx.x*blockDim.x + threadIdx.x;
  if (i >= (long long)N_EDGES*HID) return;
  int e = (int)(i >> 8);
  int d = (int)(i & 255);
  atomicAdd(&out[(size_t)dst[e]*ldo + d], h[(size_t)src[e]*HID + d]);
}

// ---------------- block-tiled WMMA GEMM with TDM-staged LDS panels ----------------
// C(M x Nout) = act(A(M x K) @ W(Nout x K)^T + bias)
// Block: 8 waves; BM=128 (wave w owns rows w*16..w*16+15), BN=64 (4 N-tiles/wave), BK=32.
// LDS: double-buffered panels, A 128x32 f32 (16KB) + W 64x32 f32 (8KB) per stage = 48KB.
#define BM 128
#define BN 64
#define BK 32
#define PANEL_F   (BM*BK + BN*BK)   // floats per stage buffer = 6144
#define A_BYTES   (BM*BK*4)         // 16384
#define STAGE_B   (PANEL_F*4)       // 24576 bytes

#ifdef HAVE_TDM
// Build a 2-D TDM descriptor and issue tensor_load_to_lds (6-arg toolchain variant).
// Group0: [1:0]=count=1 | lds_addr | global_addr | type=2.  Group1 per ISA bitfields.
__device__ __forceinline__ void tdm_load_2d(unsigned lds_off, const float* gptr,
                                            unsigned tile_w, unsigned tile_h,
                                            unsigned stride_elems, unsigned tensor_h){
  unsigned long long ga = (unsigned long long)(uintptr_t)gptr;
  u32x4 g0;
  g0[0] = 1u;                                       // count=1, user descriptor
  g0[1] = lds_off;                                  // lds_addr (bytes)
  g0[2] = (unsigned)(ga & 0xffffffffu);             // global_addr[31:0]
  g0[3] = (unsigned)((ga >> 32) & 0x01ffffffu) | (2u << 30);  // addr[56:32] | type=2
  unsigned td0 = stride_elems;                      // tensor_dim0 (row len, no OOB)
  unsigned td1 = tensor_h;                          // tensor_dim1
  unsigned long long s0 = (unsigned long long)stride_elems;
  i32x8 g1;
  g1[0] = (int)(2u << 16);                          // data_size=2 (4 bytes)
  g1[1] = (int)((td0 & 0xffffu) << 16);             // tensor_dim0[15:0] @ bits 63:48
  g1[2] = (int)((td0 >> 16) | ((td1 & 0xffffu) << 16));   // td0[31:16] | td1[15:0]
  g1[3] = (int)((td1 >> 16) | (tile_w << 16));      // td1[31:16] | tile_dim0
  g1[4] = (int)(tile_h & 0xffffu);                  // tile_dim1 (tile_dim2=0)
  g1[5] = (int)(s0 & 0xffffffffu);                  // tensor_dim0_stride[31:0]
  g1[6] = (int)((s0 >> 32) & 0xffffu);              // stride[47:32] (dim1_stride=0)
  g1[7] = 0;
  i32x4 z4 = {0, 0, 0, 0};
  i32x8 z8 = {0, 0, 0, 0, 0, 0, 0, 0};
  __builtin_amdgcn_tensor_load_to_lds(g0, g1, z4, z4, z8, 0);
}
#endif

// CDNA5 16-bit fragment layout (16x32):
//   lanes 0-15 : row = lane,    elems 0..7 = K kh+0..7, 8..15 = K 16+kh..23+kh (kh=0)
//   lanes 16-31: row = lane-16, same with kh=8
__device__ __forceinline__ v16h frag_from_lds(const float* rowp, int kh){
  v16h f;
#pragma unroll
  for (int i = 0; i < 8; ++i) f[i]     = (_Float16)rowp[kh + i];
#pragma unroll
  for (int i = 0; i < 8; ++i) f[8 + i] = (_Float16)rowp[16 + kh + i];
  return f;
}

__global__ __launch_bounds__(256) void k_gemm_wmma(
    const float* __restrict__ A, int lda,
    const float* __restrict__ W, int ldw,
    const float* __restrict__ bias,
    float* __restrict__ C, int ldc,
    int M, int Nout, int K, int relu)
{
  extern __shared__ float smem[];          // 2 stages x (A 128x32 | W 64x32) f32
  const int wave = threadIdx.x >> 5;       // wave32
  const int lane = threadIdx.x & 31;
  const int nNB  = Nout / BN;
  const int bm   = blockIdx.x / nNB;
  const int bn   = blockIdx.x - bm*nNB;
  const int m0   = bm * BM;
  const int n0   = bn * BN;
  const int rows_blk = (M - m0 < BM) ? (M - m0) : BM;
  const int rsub = lane & 15;
  const int kh   = (lane >> 4) << 3;       // 0 or 8
  const int nk   = K / BK;

  v8f acc[4] = {};

  auto stage = [&](int s, int ks){
    const int k0 = ks * BK;
#ifdef HAVE_TDM
    if (wave == 0) {
      tdm_load_2d((unsigned)(s*STAGE_B),           A + (size_t)m0*lda + k0,
                  BK, (unsigned)rows_blk, (unsigned)lda, (unsigned)M);
      tdm_load_2d((unsigned)(s*STAGE_B + A_BYTES), W + (size_t)n0*ldw + k0,
                  BK, BN, (unsigned)ldw, (unsigned)Nout);
    }
#else
    float* Ap = smem + s*PANEL_F;
    float* Wp = Ap + BM*BK;
    for (int j = threadIdx.x; j < BM*BK; j += 256) {
      int r = j >> 5, kk = j & 31;
      Ap[j] = (r < rows_blk) ? A[(size_t)(m0 + r)*lda + k0 + kk] : 0.0f;
    }
    for (int j = threadIdx.x; j < BN*BK; j += 256) {
      int r = j >> 5, kk = j & 31;
      Wp[j] = W[(size_t)(n0 + r)*ldw + k0 + kk];
    }
#endif
  };

  stage(0, 0);
  for (int i = 0; i < nk; ++i) {
    const int cur = i & 1;
    if (i + 1 < nk) stage(cur ^ 1, i + 1);
#ifdef HAVE_TDM
    if (wave == 0) {
      if (i + 1 < nk) __builtin_amdgcn_s_wait_tensorcnt(2);  // current pair done
      else            __builtin_amdgcn_s_wait_tensorcnt(0);
    }
#endif
    __syncthreads();                       // panel `cur` ready for all waves
    const float* Ap = smem + cur*PANEL_F;
    const float* Wp = Ap + BM*BK;
    v16h af = frag_from_lds(Ap + (wave*16 + rsub)*BK, kh);
#pragma unroll
    for (int nt = 0; nt < 4; ++nt) {
      v16h wf = frag_from_lds(Wp + (nt*16 + rsub)*BK, kh);
      acc[nt] = __builtin_amdgcn_wmma_f32_16x16x32_f16(false, af, false, wf,
                                                       (short)0, acc[nt], false, false);
    }
    __syncthreads();                       // done reading before next overwrite
  }

  // D layout: VGPR r -> lanes 0-15: (M=r, N=lane), lanes 16-31: (M=r+8, N=lane-16)
  if (m0 + wave*16 < M) {                  // M is a multiple of 16 -> wave-uniform guard
    const int rbase = m0 + wave*16 + ((lane >> 4) << 3);
#pragma unroll
    for (int nt = 0; nt < 4; ++nt) {
      int col = n0 + nt*16 + rsub;
      float b = bias ? bias[col] : 0.0f;
#pragma unroll
      for (int r = 0; r < 8; ++r) {
        float v = acc[nt][r] + b;
        if (relu) v = fmaxf(v, 0.0f);
        C[(size_t)(rbase + r)*ldc + col] = v;
      }
    }
  }
}

// ---------------- GRU elementwise ----------------
__global__ void k_gru(const float* __restrict__ gi, const float* __restrict__ gh,
                      const float* __restrict__ h, float* out, int n){
  int i = blockIdx.x*blockDim.x + threadIdx.x;
  if (i >= n*HID) return;
  int nd = i >> 8, c = i & 255;
  const float* gir = gi + (size_t)nd*768;
  const float* ghr = gh + (size_t)nd*768;
  float r  = sigmoidf_(gir[c]       + ghr[c]);
  float z  = sigmoidf_(gir[256 + c] + ghr[256 + c]);
  float nn = tanhf   (gir[512 + c] + r*ghr[512 + c]);
  out[i] = (1.0f - z)*nn + z*h[i];
}

// ---------------- gate alpha ----------------
__global__ void k_alpha(const float* __restrict__ g1, const float* __restrict__ w2,
                        const float* __restrict__ b2, float* alpha, int n){
  int i = blockIdx.x*blockDim.x + threadIdx.x;
  if (i >= n) return;
  float a = 0.0f;
  for (int k = 0; k < 128; ++k) a += g1[(size_t)i*128 + k]*w2[k];
  a += b2[0];
  alpha[i] = sigmoidf_(a);
}

// ---------------- masked sum ----------------
__global__ __launch_bounds__(256) void k_masked_reduce(const float* __restrict__ skipm,
                                                       const float* __restrict__ alpha,
                                                       const unsigned char* __restrict__ mask,
                                                       float* msum){
  int c  = threadIdx.x;
  int n0 = blockIdx.x * 256;
  float acc = 0.0f;
  for (int n = n0; n < n0 + 256; ++n)
    if (!mask[n]) acc += alpha[n]*skipm[(size_t)n*HID + c];
  atomicAdd(&msum[c], acc);
}

// ---------------- hh tail rows ----------------
__global__ void k_set_tail(float* hh, const float* rc_init, const float* s_init){
  int i = blockIdx.x*blockDim.x + threadIdx.x;
  if (i >= (NPAD - N_NODES)*HID) return;
  int r = N_NODES + (i >> 8), c = i & 255;
  float v = 0.0f;
  if (r == N_NODES)          v = rc_init[c];
  else if (r == N_NODES + 1) v = s_init[c];
  hh[(size_t)r*HID + c] = v;
}

// ---------------- super-node GRU (1 row) ----------------
__global__ __launch_bounds__(256) void k_super_gru(
    const float* __restrict__ msum, const float* __restrict__ rc_init,
    const float* __restrict__ skip_W, const float* __restrict__ s_init,
    const float* __restrict__ wih, const float* __restrict__ whh,
    const float* __restrict__ bih, const float* __restrict__ bhh, float* hh){
  __shared__ float mt[HID];
  int c = threadIdx.x;
  float mrc = 0.0f;
  for (int k = 0; k < HID; ++k) mrc += rc_init[k]*skip_W[(size_t)c*HID + k];
  mt[c] = msum[c] + mrc;
  __syncthreads();
  float gi0 = bih[c], gi1 = bih[256 + c], gi2 = bih[512 + c];
  float gh0 = bhh[c], gh1 = bhh[256 + c], gh2 = bhh[512 + c];
  for (int k = 0; k < HID; ++k) {
    float mk = mt[k], sk = s_init[k];
    gi0 += mk*wih[(size_t)c*HID + k];
    gi1 += mk*wih[(size_t)(256 + c)*HID + k];
    gi2 += mk*wih[(size_t)(512 + c)*HID + k];
    gh0 += sk*whh[(size_t)c*HID + k];
    gh1 += sk*whh[(size_t)(256 + c)*HID + k];
    gh2 += sk*whh[(size_t)(512 + c)*HID + k];
  }
  float r  = sigmoidf_(gi0 + gh0);
  float z  = sigmoidf_(gi1 + gh1);
  float nn = tanhf(gi2 + r*gh2);
  hh[(size_t)(N_NODES + 1)*HID + c] = (1.0f - z)*nn + z*s_init[c];
}

// ---------------- single-row attention + out proj + JK max ----------------
__global__ __launch_bounds__(256) void k_attn_tail(
    const float* __restrict__ hh, const float* __restrict__ q_w, const float* __restrict__ q_b,
    const float* __restrict__ Kb, const float* __restrict__ Vb,
    const float* __restrict__ o_w, const float* __restrict__ o_b,
    float* scores, float* out){
  __shared__ float q[HID];
  __shared__ float red[256];
  __shared__ float ctx[HID];
  int c = threadIdx.x;
  const float* hs = hh + (size_t)(N_NODES + 1)*HID;
  float qv = q_b[c];
  for (int k = 0; k < HID; ++k) qv += hs[k]*q_w[(size_t)c*HID + k];
  q[c] = qv;
  __syncthreads();
  float lmax = -1e30f;
  for (int i = c; i < NN; i += 256) {
    const float* kr = Kb + (size_t)i*HID;
    float s = 0.0f;
    for (int k = 0; k < HID; ++k) s += q[k]*kr[k];
    s *= 0.0625f;                       // 1/sqrt(256)
    scores[i] = s;
    lmax = fmaxf(lmax, s);
  }
  red[c] = lmax; __syncthreads();
  for (int st = 128; st > 0; st >>= 1) { if (c < st) red[c] = fmaxf(red[c], red[c + st]); __syncthreads(); }
  float mx = red[0]; __syncthreads();
  float lsum = 0.0f;
  for (int i = c; i < NN; i += 256) {
    float e = __expf(scores[i] - mx);
    scores[i] = e;
    lsum += e;
  }
  red[c] = lsum; __syncthreads();
  for (int st = 128; st > 0; st >>= 1) { if (c < st) red[c] += red[c + st]; __syncthreads(); }
  float tot = red[0]; __syncthreads();
  float cc = 0.0f;
  for (int i = 0; i < NN; ++i) cc += scores[i]*Vb[(size_t)i*HID + c];
  ctx[c] = cc / tot;
  __syncthreads();
  float ov = o_b[c];
  for (int k = 0; k < HID; ++k) ov += ctx[k]*o_w[(size_t)c*HID + k];
  out[c] = fmaxf(0.0f, hs[c] + ov);    // h_pad[N+1] == 0
}

// ==================== host orchestration ====================
extern "C" void kernel_launch(void* const* d_in, const int* in_sizes, int n_in,
                              void* d_out, int out_size, void* d_ws, size_t ws_size,
                              hipStream_t stream) {
  (void)in_sizes; (void)n_in; (void)out_size; (void)ws_size;
  const float* x         = (const float*)d_in[0];
  const int*   ei        = (const int*)d_in[1];
  const int*   src       = ei;
  const int*   dst       = ei + N_EDGES;
  const float* edge_attr = (const float*)d_in[2];
  const unsigned char* rc_mask = (const unsigned char*)d_in[3];
  const float* W_msg1   = (const float*)d_in[4];
  const float* g1_wih   = (const float*)d_in[5];
  const float* g1_whh   = (const float*)d_in[6];
  const float* g1_bih   = (const float*)d_in[7];
  const float* g1_bhh   = (const float*)d_in[8];
  const float* W_msg2   = (const float*)d_in[9];
  const float* g2_wih   = (const float*)d_in[10];
  const float* g2_whh   = (const float*)d_in[11];
  const float* g2_bih   = (const float*)d_in[12];
  const float* g2_bhh   = (const float*)d_in[13];
  const float* gin_w1   = (const float*)d_in[14];
  const float* gin_b1   = (const float*)d_in[15];
  const float* gin_w2   = (const float*)d_in[16];
  const float* gin_b2   = (const float*)d_in[17];
  const float* proj_w   = (const float*)d_in[18];
  const float* proj_b   = (const float*)d_in[19];
  const float* q_w      = (const float*)d_in[20];
  const float* q_b      = (const float*)d_in[21];
  const float* k_w      = (const float*)d_in[22];
  const float* k_b      = (const float*)d_in[23];
  const float* v_w      = (const float*)d_in[24];
  const float* v_b      = (const float*)d_in[25];
  const float* o_w      = (const float*)d_in[26];
  const float* o_b      = (const float*)d_in[27];
  const float* gate_w1  = (const float*)d_in[28];
  const float* gate_b1  = (const float*)d_in[29];
  const float* gate_w2  = (const float*)d_in[30];
  const float* gate_b2  = (const float*)d_in[31];
  const float* skip_W   = (const float*)d_in[32];
  const float* sg_wih   = (const float*)d_in[33];
  const float* sg_whh   = (const float*)d_in[34];
  const float* sg_bih   = (const float*)d_in[35];
  const float* sg_bhh   = (const float*)d_in[36];
  const float* rc_init  = (const float*)d_in[37];
  const float* s_init   = (const float*)d_in[38];

  // ---- workspace (floats); ~110 MB ----
  float* ws = (float*)d_ws;
  size_t off = 0;
  auto alloc = [&](size_t n){ float* p = ws + off; off += n; return p; };
  float* gi    = alloc((size_t)N_NODES*768);   // phase2: cat3; phase3: K/V
  float* gh    = alloc((size_t)N_NODES*768);   // phase2: chainA/chainB
  float* S     = alloc((size_t)N_NODES*CATK);  // scatter buffer / GIN agg
  float* mbuf  = alloc((size_t)N_NODES*HID);
  float* hA    = alloc((size_t)N_NODES*HID);
  float* hB    = alloc((size_t)N_NODES*HID);
  float* SE    = alloc((size_t)N_NODES*ED);
  float* hh    = alloc((size_t)NPAD*HID);
  float* gbuf  = alloc((size_t)N_NODES*128);
  float* alpha = alloc((size_t)N_NODES);
  float* skipm = alloc((size_t)N_NODES*HID);
  float* msum  = alloc(256);
  float* scores= alloc(NPAD);
  float* cat3   = gi;
  float* chainA = gh;
  float* chainB = gh + (size_t)N_NODES*HID;
  float* Kbuf   = gi;
  float* Vbuf   = gi + (size_t)NPAD*HID;

  auto zero = [&](float* p, size_t n){
    k_zero<<<(unsigned)((n + 255)/256), 256, 0, stream>>>(p, (int)n);
  };
  auto gemm = [&](const float* A, int lda, const float* W, int ldw, const float* b,
                  float* C, int ldc, int M, int Nout, int K, int relu){
    int blocks = ((M + BM - 1)/BM) * (Nout/BN);
    k_gemm_wmma<<<blocks, 256, 2*PANEL_F*4, stream>>>(A, lda, W, ldw, b, C, ldc,
                                                      M, Nout, K, relu);
  };
  auto scatterN = [&](const float* hin, float* outp, int ldo){
    long long tot = (long long)N_EDGES*HID;
    k_scatter_node256<<<(unsigned)((tot + 255)/256), 256, 0, stream>>>(hin, src, dst, outp, ldo);
  };

  // edge_attr segment_sum once (layer-invariant)
  zero(SE, (size_t)N_NODES*ED);
  {
    long long tot = (long long)N_EDGES*ED;
    k_scatter_edge64<<<(unsigned)((tot + 255)/256), 256, 0, stream>>>(edge_attr, dst, SE, ED);
  }

  auto dmpnn = [&](const float* hin, const float* Wm, const float* wih, const float* whh,
                   const float* bih, const float* bhh, float* hout){
    zero(S, (size_t)N_NODES*CATK);
    scatterN(hin, S, CATK);
    k_copy_cols<<<(N_NODES*ED + 255)/256, 256, 0, stream>>>(SE, ED, S, CATK, HID, N_NODES, ED);
    gemm(S,    CATK, Wm,  CATK, nullptr, mbuf, HID, N_NODES, HID, CATK, 0);
    gemm(mbuf, HID,  wih, HID,  bih,     gi,   768, N_NODES, 768, HID,  0);
    gemm(hin,  HID,  whh, HID,  bhh,     gh,   768, N_NODES, 768, HID,  0);
    k_gru<<<(N_NODES*HID + 255)/256, 256, 0, stream>>>(gi, gh, hin, hout, N_NODES);
  };
  dmpnn(x,  W_msg1, g1_wih, g1_whh, g1_bih, g1_bhh, hA);
  dmpnn(hA, W_msg2, g2_wih, g2_whh, g2_bih, g2_bhh, hB);

  auto gin = [&](const float* hin, int wi, float* outp, int ldo){
    const float* w1 = gin_w1 + (size_t)wi*HID*HID;
    const float* b1 = gin_b1 + (size_t)wi*HID;
    const float* w2 = gin_w2 + (size_t)wi*HID*HID;
    const float* b2 = gin_b2 + (size_t)wi*HID;
    zero(S, (size_t)N_NODES*HID);
    scatterN(hin, S, HID);
    k_add<<<(N_NODES*HID + 255)/256, 256, 0, stream>>>(hin, S, mbuf, N_NODES*HID);
    gemm(mbuf, HID, w1, HID, b1, hA,   HID, N_NODES, HID, HID, 1);
    gemm(hA,   HID, w2, HID, b2, outp, ldo, N_NODES, HID, HID, 0);
  };
  gin(hB,     0, cat3,       768);
  gin(hB,     1, chainA,     256);
  gin(chainA, 1, cat3 + 256, 768);
  gin(hB,     2, chainA,     256);
  gin(chainA, 2, chainB,     256);
  gin(chainB, 2, cat3 + 512, 768);

  gemm(cat3, 768, proj_w, 768, proj_b, hh, HID, N_NODES, HID, 768, 0);
  k_set_tail<<<16, 256, 0, stream>>>(hh, rc_init, s_init);

  gemm(hh, HID, gate_w1, HID, gate_b1, gbuf, 128, N_NODES, 128, HID, 1);
  k_alpha<<<N_NODES/256, 256, 0, stream>>>(gbuf, gate_w2, gate_b2, alpha, N_NODES);
  gemm(hh, HID, skip_W, HID, nullptr, skipm, HID, N_NODES, HID, HID, 0);
  zero(msum, 256);
  k_masked_reduce<<<N_NODES/256, 256, 0, stream>>>(skipm, alpha, rc_mask, msum);
  k_super_gru<<<1, 256, 0, stream>>>(msum, rc_init, skip_W, s_init,
                                     sg_wih, sg_whh, sg_bih, sg_bhh, hh);

  gemm(hh, HID, k_w, HID, k_b, Kbuf, HID, NPAD, HID, HID, 0);
  gemm(hh, HID, v_w, HID, v_b, Vbuf, HID, NPAD, HID, HID, 0);
  k_attn_tail<<<1, 256, 0, stream>>>(hh, q_w, q_b, Kbuf, Vbuf, o_w, o_b,
                                     scores, (float*)d_out);
}
